// Block_58652073394865
// MI455X (gfx1250) — matrix operand order repaired and checked by reading, not compiled, not run on hardware
//
#include <hip/hip_runtime.h>
#include <math.h>

// Problem dims (fixed by the reference)
#define BB 16
#define SS 577
#define DD 1024
#define HH 16
#define FF 4096
#define DHH 64
#define MROWS (BB * SS)   // 9232
#define SP 608            // 577 padded to multiple of 32 (19 chunks)
#define NT 37             // ceil(577/16) key tiles
#define GPAD 40           // LDS row pitch (elements) for 32-elem k-slabs

typedef __attribute__((ext_vector_type(16))) __bf16 bf16x16;
typedef __attribute__((ext_vector_type(8)))  float  floatx8;

__device__ __forceinline__ floatx8 wmma_bf16(bf16x16 a, bf16x16 b, floatx8 c) {
  // (neg_a, A, neg_b, B, c_mod, C, reuse_a, reuse_b)
  return __builtin_amdgcn_wmma_f32_16x16x32_bf16(false, a, false, b, (short)0, c,
                                                 false, false);
}

// Generic pointer -> wave-relative LDS byte offset (LDS aperture keeps the
// offset in addr[31:0]).
__device__ __forceinline__ unsigned lds_off(const void* p) {
  return (unsigned)(uintptr_t)p;
}

// Async 16B global -> LDS copy (per lane), tracked by ASYNCcnt.
__device__ __forceinline__ void async_ld_b128(unsigned lds_byte_off,
                                              const void* gaddr) {
  asm volatile("global_load_async_to_lds_b128 %0, %1, off"
               :: "v"(lds_byte_off),
                  "v"((unsigned long long)(uintptr_t)gaddr)
               : "memory");
}

#define WAIT_ASYNC(n) asm volatile("s_wait_asynccnt " #n ::: "memory")

// ---------------------------------------------------------------------------
// Weight f32 [K x N] -> bf16 transposed [N x K], LDS-tiled 32x32.
// K, N multiples of 32. block (32,8), grid (N/32, K/32).
// ---------------------------------------------------------------------------
__global__ __launch_bounds__(256)
void transpose_w_bf16_kernel(const float* __restrict__ src,
                             __bf16* __restrict__ dst, int K, int N) {
  __shared__ __bf16 t[32][33];
  const int n0 = blockIdx.x * 32;
  const int k0 = blockIdx.y * 32;
  for (int dy = threadIdx.y; dy < 32; dy += 8) {
    t[dy][threadIdx.x] = (__bf16)src[(size_t)(k0 + dy) * N + n0 + threadIdx.x];
  }
  __syncthreads();
  for (int dy = threadIdx.y; dy < 32; dy += 8) {
    dst[(size_t)(n0 + dy) * K + k0 + threadIdx.x] = t[threadIdx.x][dy];
  }
}

// ---------------------------------------------------------------------------
// V bf16 [B*S, D] -> Vt bf16 [B][H][DH][SP], zero-padded for s >= S.
// block (32,8), grid (SP/32=19, DH/32=2, B*H).
// ---------------------------------------------------------------------------
__global__ __launch_bounds__(256)
void transpose_v_kernel(const __bf16* __restrict__ v, __bf16* __restrict__ vt) {
  __shared__ __bf16 t[32][33];
  const int bh = blockIdx.z;
  const int b = bh >> 4, h = bh & 15;
  const int s0 = blockIdx.x * 32;
  const int d0 = blockIdx.y * 32;
  for (int dy = threadIdx.y; dy < 32; dy += 8) {
    const int s1 = s0 + dy;
    __bf16 val = (__bf16)0.0f;
    if (s1 < SS) val = v[((size_t)b * SS + s1) * DD + h * DHH + d0 + threadIdx.x];
    t[dy][threadIdx.x] = val;
  }
  __syncthreads();
  for (int dy = threadIdx.y; dy < 32; dy += 8) {
    const int d = d0 + dy;
    vt[(((size_t)b * HH + h) * DHH + d) * SP + s0 + threadIdx.x] = t[threadIdx.x][dy];
  }
}

// ---------------------------------------------------------------------------
// LayerNorm over last dim (D=1024), f32 in -> bf16 out. One block per row.
// ---------------------------------------------------------------------------
__global__ __launch_bounds__(256)
void layernorm_bf16_kernel(const float* __restrict__ x,
                           const float* __restrict__ g,
                           const float* __restrict__ bta,
                           __bf16* __restrict__ out) {
  __shared__ float rs[256], rs2[256];
  const int row = blockIdx.x;
  const float* xr = x + (size_t)row * DD;
  float s = 0.f, s2 = 0.f;
  for (int j = threadIdx.x; j < DD; j += 256) {
    float v = xr[j];
    s += v; s2 += v * v;
  }
  rs[threadIdx.x] = s; rs2[threadIdx.x] = s2;
  __syncthreads();
  for (int st = 128; st > 0; st >>= 1) {
    if (threadIdx.x < st) {
      rs[threadIdx.x]  += rs[threadIdx.x + st];
      rs2[threadIdx.x] += rs2[threadIdx.x + st];
    }
    __syncthreads();
  }
  const float mu  = rs[0] * (1.0f / DD);
  const float var = rs2[0] * (1.0f / DD) - mu * mu;
  const float inv = rsqrtf(var + 1e-6f);
  __bf16* orow = out + (size_t)row * DD;
  for (int j = threadIdx.x; j < DD; j += 256)
    orow[j] = (__bf16)((xr[j] - mu) * inv * g[j] + bta[j]);
}

// ---------------------------------------------------------------------------
// bf16 WMMA GEMM with async-LDS double-buffered staging:
//   out = act(A[MxK] @ B[KxN] + bias) (+ resid)
// A row-major bf16 (lda=K). B supplied TRANSPOSED: Bt[N x K] row-major.
// Block tile 64(M) x 256(N), k-step 32. Per k-step the block stages
//   A-slab 64x32 and B-slab 256x32 into LDS once (global_load_async_to_lds),
// then 8 waves (2M x 4N, 32x64 tile each) read fragments as ds_load_b128
// and issue 8 WMMA each. 2-deep pipeline via s_wait_asynccnt + barriers.
// K % 32 == 0, N % 256 == 0 assumed; M guarded.
// act: 0=none, 1=exact GELU. outF / outB / resid nullable.
// ---------------------------------------------------------------------------
__global__ __launch_bounds__(256)
void gemm_bf16_kernel(const __bf16* __restrict__ A,
                      const __bf16* __restrict__ Bt,
                      const float* __restrict__ bias,
                      const float* __restrict__ resid,
                      float* __restrict__ outF,
                      __bf16* __restrict__ outB,
                      int M, int N, int K, int act) {
  __shared__ __bf16 As[2][64 * GPAD];    // 2 x 5 KB
  __shared__ __bf16 Bs[2][256 * GPAD];   // 2 x 20 KB

  const int tid  = threadIdx.x;
  const int wave = tid >> 5;
  const int lane = tid & 31;
  const int tileM = blockIdx.x * 64;
  const int tileN = blockIdx.y * 256;
  const int wm = (wave >> 2) * 32;       // wave M offset in block tile
  const int wn = (wave & 3) * 64;        // wave N offset in block tile
  const int ml = lane & 15;
  const int lh = lane >> 4;

  // ---- cooperative staging addresses (one A chunk + four B chunks/thread) --
  const int aRow = tid >> 2;
  const int aCh  = (tid & 3) * 8;
  int aSrcRow = tileM + aRow; if (aSrcRow >= M) aSrcRow = M - 1;
  const __bf16* aSrcBase = A + (size_t)aSrcRow * K + aCh;
  const unsigned aLds = (unsigned)((aRow * GPAD + aCh) * 2);

  const __bf16* bSrcBase[4];
  unsigned bLds[4];
  #pragma unroll
  for (int j = 0; j < 4; ++j) {
    const int idx  = tid + j * 256;
    const int bRow = idx >> 2;
    const int bCh  = (idx & 3) * 8;
    bSrcBase[j] = Bt + (size_t)(tileN + bRow) * K + bCh;
    bLds[j] = (unsigned)((bRow * GPAD + bCh) * 2);
  }
  const unsigned asBase[2] = {lds_off(&As[0][0]), lds_off(&As[1][0])};
  const unsigned bsBase[2] = {lds_off(&Bs[0][0]), lds_off(&Bs[1][0])};

  auto issue_stage = [&](int kk, int buf) {
    async_ld_b128(asBase[buf] + aLds, aSrcBase + kk);
    #pragma unroll
    for (int j = 0; j < 4; ++j)
      async_ld_b128(bsBase[buf] + bLds[j], bSrcBase[j] + kk);
  };

  floatx8 cc[2][4];
  #pragma unroll
  for (int m2 = 0; m2 < 2; ++m2)
    #pragma unroll
    for (int s = 0; s < 4; ++s) cc[m2][s] = (floatx8){0, 0, 0, 0, 0, 0, 0, 0};

  const int nsteps = K / 32;
  issue_stage(0, 0);

  for (int i = 0; i < nsteps; ++i) {
    const int buf = i & 1;
    if (i + 1 < nsteps) {
      issue_stage((i + 1) * 32, buf ^ 1);
      WAIT_ASYNC(0x5);                 // keep only the 5 just-issued in flight
    } else {
      WAIT_ASYNC(0x0);
    }
    __syncthreads();                   // all waves' stage-i data in LDS

    const __bf16* as = &As[buf][0];
    const __bf16* bs = &Bs[buf][0];

    bf16x16 a0, a1;
    #pragma unroll
    for (int t = 0; t < 16; ++t) {
      const int off = lh * 16 + t;
      a0[t] = as[(wm + ml) * GPAD + off];
      a1[t] = as[(wm + 16 + ml) * GPAD + off];
    }
    bf16x16 bb[4];
    #pragma unroll
    for (int s = 0; s < 4; ++s) {
      const __bf16* r = bs + (wn + s * 16 + ml) * GPAD + lh * 16;
      #pragma unroll
      for (int t = 0; t < 16; ++t) bb[s][t] = r[t];
    }
    #pragma unroll
    for (int s = 0; s < 4; ++s) {
      cc[0][s] = wmma_bf16(a0, bb[s], cc[0][s]);
      cc[1][s] = wmma_bf16(a1, bb[s], cc[1][s]);
    }
    __syncthreads();                   // reads done before buffer is re-filled
  }

  // Epilogue: C layout row = m2*16 + lh*8 + r, col = ml
  #pragma unroll
  for (int m2 = 0; m2 < 2; ++m2) {
    #pragma unroll
    for (int s = 0; s < 4; ++s) {
      const int col = tileN + wn + s * 16 + ml;
      #pragma unroll
      for (int r = 0; r < 8; ++r) {
        const int row = tileM + wm + m2 * 16 + lh * 8 + r;
        if (row < M) {
          float v = cc[m2][s][r] + bias[col];
          if (act == 1) v = 0.5f * v * (1.0f + erff(v * 0.70710678118654752f));
          if (resid) v += resid[(size_t)row * N + col];
          if (outF) outF[(size_t)row * N + col] = v;
          if (outB) outB[(size_t)row * N + col] = (__bf16)v;
        }
      }
    }
  }
}

// ---------------------------------------------------------------------------
// Attention: one wave per (q-tile of 16, head, batch).
// Q/K/ctx are bf16 [B*S, D]; head h lives in columns h*64..h*64+63.
// Vt is bf16 [B][H][DH][SP] (pre-transposed, zero-padded keys).
// Scores kept in LDS (16 x 608 f32), softmax f32, probs repacked bf16 in LDS.
// ---------------------------------------------------------------------------
__global__ __launch_bounds__(32)
void attn_kernel(const __bf16* __restrict__ Q,
                 const __bf16* __restrict__ Kmat,
                 const __bf16* __restrict__ Vt,
                 __bf16* __restrict__ ctx) {
  __shared__ float  sc[16][SP];   // raw scores / exp values
  __shared__ __bf16 pb[16][SP];   // normalized probs (bf16)

  const int lane = threadIdx.x;
  const int qt = blockIdx.x, h = blockIdx.y, bidx = blockIdx.z;
  const int ml = lane & 15;
  const int lh = lane >> 4;
  const int hd = h * DHH;
  const size_t rowbase = (size_t)bidx * SS;

  int qrow = qt * 16 + ml;
  if (qrow >= SS) qrow = SS - 1;
  const __bf16* Qp = Q + (rowbase + qrow) * DD + hd;

  // Preload Q A-fragments for the two k=32 chunks of DH=64
  bf16x16 aq0, aq1;
  #pragma unroll
  for (int i = 0; i < 16; ++i) {
    const int kidx = lh * 16 + i;
    aq0[i] = Qp[kidx];
    aq1[i] = Qp[32 + kidx];
  }

  // ---- Phase 1: scores = (Q K^T) * 1/sqrt(64) into LDS ----
  for (int t = 0; t < NT; ++t) {
    int krow = t * 16 + ml;
    if (krow >= SS) krow = SS - 1;
    const __bf16* Kp = Kmat + (rowbase + krow) * DD + hd + lh * 16;
    bf16x16 b0, b1;
    #pragma unroll
    for (int i = 0; i < 16; ++i) {
      b0[i] = Kp[i];         // dh = lh*16 + i
      b1[i] = Kp[32 + i];    // dh = 32 + lh*16 + i
    }
    floatx8 s = (floatx8){0, 0, 0, 0, 0, 0, 0, 0};
    s = wmma_bf16(aq0, b0, s);
    s = wmma_bf16(aq1, b1, s);
    #pragma unroll
    for (int r = 0; r < 8; ++r)
      sc[lh * 8 + r][t * 16 + ml] = s[r] * 0.125f;
  }
  __syncthreads();

  // ---- Phase 2: softmax. Row = ml, two lanes (lh=0/1) split columns. ----
  {
    const int row = ml, off = lh;
    float mx = -3.0e38f;
    for (int j = off; j < SS; j += 2) mx = fmaxf(mx, sc[row][j]);
    mx = fmaxf(mx, __shfl_xor(mx, 16));
    float sum = 0.f;
    for (int j = off; j < SS; j += 2) {
      float e = __expf(sc[row][j] - mx);
      sc[row][j] = e;
      sum += e;
    }
    sum += __shfl_xor(sum, 16);
    const float inv = 1.0f / sum;
    for (int j = off; j < SS; j += 2) pb[row][j] = (__bf16)(sc[row][j] * inv);
    for (int j = SS + off; j < SP; j += 2) pb[row][j] = (__bf16)0.0f;
  }
  __syncthreads();

  // ---- Phase 3: ctx = P @ V  (19 chunks of 32 keys, 4 dh-subtiles) ----
  floatx8 cc[4];
  #pragma unroll
  for (int s4 = 0; s4 < 4; ++s4) cc[s4] = (floatx8){0, 0, 0, 0, 0, 0, 0, 0};

  // Vt fragment base: column d = s4*16 + ml, keys contiguous per lane
  const __bf16* Vp = Vt + (((size_t)bidx * HH + h) * DHH + ml) * SP + lh * 16;

  for (int kc = 0; kc < SP; kc += 32) {
    bf16x16 a;
    #pragma unroll
    for (int i = 0; i < 16; ++i) {
      const int kidx = lh * 16 + i;
      a[i] = pb[ml][kc + kidx];
    }
    #pragma unroll
    for (int s4 = 0; s4 < 4; ++s4) {
      const __bf16* r = Vp + (size_t)s4 * 16 * SP + kc;
      bf16x16 b;
      #pragma unroll
      for (int i = 0; i < 16; ++i) b[i] = r[i];
      cc[s4] = wmma_bf16(a, b, cc[s4]);
    }
  }

  // Store ctx (bf16, [B*S, D] with head-interleaved columns)
  #pragma unroll
  for (int s4 = 0; s4 < 4; ++s4) {
    #pragma unroll
    for (int r = 0; r < 8; ++r) {
      const int qr = qt * 16 + lh * 8 + r;
      if (qr < SS)
        ctx[(rowbase + qr) * DD + hd + s4 * 16 + ml] = (__bf16)cc[s4][r];
    }
  }
}

// ---------------------------------------------------------------------------
// Host-side orchestration
// ---------------------------------------------------------------------------
extern "C" void kernel_launch(void* const* d_in, const int* in_sizes, int n_in,
                              void* d_out, int out_size, void* d_ws, size_t ws_size,
                              hipStream_t stream) {
  (void)in_sizes; (void)n_in; (void)out_size; (void)ws_size;
  const float* x     = (const float*)d_in[0];
  const float* ln1_g = (const float*)d_in[1];
  const float* ln1_b = (const float*)d_in[2];
  const float* wq    = (const float*)d_in[3];
  const float* bq    = (const float*)d_in[4];
  const float* wk    = (const float*)d_in[5];
  const float* bk    = (const float*)d_in[6];
  const float* wv    = (const float*)d_in[7];
  const float* bv    = (const float*)d_in[8];
  const float* wo    = (const float*)d_in[9];
  const float* bo    = (const float*)d_in[10];
  const float* ln2_g = (const float*)d_in[11];
  const float* ln2_b = (const float*)d_in[12];
  const float* w1    = (const float*)d_in[13];
  const float* b1    = (const float*)d_in[14];
  const float* w2    = (const float*)d_in[15];
  const float* b2    = (const float*)d_in[16];
  float* out = (float*)d_out;

  // Workspace layout
  char* p = (char*)d_ws;
  auto alloc = [&](size_t bytes) -> char* {
    char* r = p;
    p += (bytes + 255) & ~(size_t)255;
    return r;
  };
  __bf16* wqb  = (__bf16*)alloc((size_t)DD * DD * 2);   // [N=D x K=D]
  __bf16* wkb  = (__bf16*)alloc((size_t)DD * DD * 2);
  __bf16* wvb  = (__bf16*)alloc((size_t)DD * DD * 2);
  __bf16* wob  = (__bf16*)alloc((size_t)DD * DD * 2);
  __bf16* w1b  = (__bf16*)alloc((size_t)FF * DD * 2);   // [N=F x K=D]
  __bf16* w2b  = (__bf16*)alloc((size_t)DD * FF * 2);   // [N=D x K=F]
  __bf16* xn   = (__bf16*)alloc((size_t)MROWS * DD * 2);
  __bf16* qb   = (__bf16*)alloc((size_t)MROWS * DD * 2);
  __bf16* kb   = (__bf16*)alloc((size_t)MROWS * DD * 2);
  __bf16* vb   = (__bf16*)alloc((size_t)MROWS * DD * 2);
  __bf16* vt   = (__bf16*)alloc((size_t)BB * HH * DHH * SP * 2);
  __bf16* ctxb = (__bf16*)alloc((size_t)MROWS * DD * 2);
  float*  x1   = (float*)alloc((size_t)MROWS * DD * 4);
  __bf16* xn2  = (__bf16*)alloc((size_t)MROWS * DD * 2);
  __bf16* hmid = (__bf16*)alloc((size_t)MROWS * FF * 2);

  const int MT = (MROWS + 63) / 64;  // 145 M-tiles
  const dim3 tb(32, 8);

  // Weights -> bf16, transposed to [N x K]
  transpose_w_bf16_kernel<<<dim3(DD / 32, DD / 32), tb, 0, stream>>>(wq, wqb, DD, DD);
  transpose_w_bf16_kernel<<<dim3(DD / 32, DD / 32), tb, 0, stream>>>(wk, wkb, DD, DD);
  transpose_w_bf16_kernel<<<dim3(DD / 32, DD / 32), tb, 0, stream>>>(wv, wvb, DD, DD);
  transpose_w_bf16_kernel<<<dim3(DD / 32, DD / 32), tb, 0, stream>>>(wo, wob, DD, DD);
  transpose_w_bf16_kernel<<<dim3(FF / 32, DD / 32), tb, 0, stream>>>(w1, w1b, DD, FF);
  transpose_w_bf16_kernel<<<dim3(DD / 32, FF / 32), tb, 0, stream>>>(w2, w2b, FF, DD);

  // LN1
  layernorm_bf16_kernel<<<MROWS, 256, 0, stream>>>(x, ln1_g, ln1_b, xn);

  // Q, K, V projections (bf16 out)
  gemm_bf16_kernel<<<dim3(MT, DD / 256), 256, 0, stream>>>(
      xn, wqb, bq, nullptr, nullptr, qb, MROWS, DD, DD, 0);
  gemm_bf16_kernel<<<dim3(MT, DD / 256), 256, 0, stream>>>(
      xn, wkb, bk, nullptr, nullptr, kb, MROWS, DD, DD, 0);
  gemm_bf16_kernel<<<dim3(MT, DD / 256), 256, 0, stream>>>(
      xn, wvb, bv, nullptr, nullptr, vb, MROWS, DD, DD, 0);

  // V -> per-head transposed, zero-padded
  transpose_v_kernel<<<dim3(SP / 32, DHH / 32, BB * HH), tb, 0, stream>>>(vb, vt);

  // Attention
  attn_kernel<<<dim3(NT, HH, BB), 32, 0, stream>>>(qb, kb, vt, ctxb);

  // Output projection + residual -> x1 (f32)
  gemm_bf16_kernel<<<dim3(MT, DD / 256), 256, 0, stream>>>(
      ctxb, wob, bo, x, x1, nullptr, MROWS, DD, DD, 0);

  // LN2
  layernorm_bf16_kernel<<<MROWS, 256, 0, stream>>>(x1, ln2_g, ln2_b, xn2);

  // MLP up + GELU (bf16 out)
  gemm_bf16_kernel<<<dim3(MT, FF / 256), 256, 0, stream>>>(
      xn2, w1b, b1, nullptr, nullptr, hmid, MROWS, FF, DD, 1);

  // MLP down + residual -> d_out (f32)
  gemm_bf16_kernel<<<dim3(MT, DD / 256), 256, 0, stream>>>(
      hmid, w2b, b2, x1, out, nullptr, MROWS, DD, FF, 0);
}
